// AttentionLayer_89455578841655
// MI455X (gfx1250) — matrix operand order repaired and checked by reading, not compiled, not run on hardware
//
#include <hip/hip_runtime.h>
#include <hip/hip_bf16.h>

// ---------------------------------------------------------------------------
// Problem constants (from reference)
// ---------------------------------------------------------------------------
#define S_LEN  2048
#define HID    4096
#define NH     32
#define NKV    8
#define DHEAD  128
#define QKV_N  ((NH + 2 * NKV) * DHEAD)   // 6144
#define EPSF   1e-6f
#define LN_THETA 9.210340371976184f       // ln(10000)

typedef __bf16 bf16;
typedef __attribute__((ext_vector_type(16))) __bf16 v16bf;
typedef __attribute__((ext_vector_type(8)))  float  v8f;

__device__ __forceinline__ v8f vzero() {
  v8f z = {0.f, 0.f, 0.f, 0.f, 0.f, 0.f, 0.f, 0.f};
  return z;
}

__device__ __forceinline__ v8f wmma_bf16(v16bf a, v16bf b, v8f c) {
  // D = A(16x32 bf16) * B(32x16 bf16) + C(16x16 f32)
  return __builtin_amdgcn_wmma_f32_16x16x32_bf16(
      /*neg_a=*/false, a, /*neg_b=*/false, b,
      /*c_mod=*/(short)0, c, /*reuse_a=*/false, /*reuse_b=*/false);
}

__device__ __forceinline__ bf16 us2bf(unsigned short u) {
  return __builtin_bit_cast(bf16, u);
}
__device__ __forceinline__ unsigned int pk2(float x, float y) {
  unsigned short lo = __builtin_bit_cast(unsigned short, (bf16)x);
  unsigned short hi = __builtin_bit_cast(unsigned short, (bf16)y);
  return (unsigned int)lo | ((unsigned int)hi << 16);
}

// Async global -> LDS copy, 16B per lane (VGLOBAL GLOBAL_LOAD_ASYNC_TO_LDS).
// Tracked with ASYNCcnt; LDS address = low 32 bits of the generic pointer
// (flat addressing rule).
__device__ __forceinline__ void async_g2l_b128(void* lds_dst, const void* gsrc) {
  unsigned lds_off = (unsigned)(unsigned long long)lds_dst;
  asm volatile("global_load_async_to_lds_b128 %0, %1, off"
               :: "v"(lds_off), "v"(gsrc)
               : "memory");
}
__device__ __forceinline__ void wait_asynccnt0() {
  asm volatile("s_wait_asynccnt 0" ::: "memory");
}

// A-fragment (16x32 bf16): lane = M row (lo/hi half-wave picks
// K 0..15 / 16..31); VGPR r holds K pair r*2.
// rowp must already include the per-lane (lane>>4)*16 offset.
struct U32x8 { unsigned int v[8]; };
__device__ __forceinline__ v16bf load_a_frag(const bf16* rowp) {
  U32x8 t;
#pragma unroll
  for (int r = 0; r < 8; ++r) {
    int k = r * 2;
    t.v[r] = *(const unsigned int*)(rowp + k);
  }
  return __builtin_bit_cast(v16bf, t);
}

// B-fragment (32x16 bf16, same K order as A-frag): lane = N col (lo/hi half-wave picks
// K 0..15 / 16..31); elements are 16 contiguous K values. p must already
// include the per-lane (lane>>4)*16 offset and be 16B aligned.
struct U4x2 { uint4 a, b; };
__device__ __forceinline__ v16bf load_b_frag(const bf16* p) {
  U4x2 t;
  t.a = *(const uint4*)p;
  t.b = *(const uint4*)(p + 8);
  return __builtin_bit_cast(v16bf, t);
}

// ---------------------------------------------------------------------------
// Kernel 1/4: GEMM  C[M,N](f32) = A[M,K](f32) * B[N,K](f32)^T  via bf16 WMMA
// Block: 256 threads (8 waves), tile 128x128, K-step 32.
// Waves in 4(M) x 2(N): each wave computes 32x64 (2x4 accumulators).
// ---------------------------------------------------------------------------
#define GSTR 40   // LDS row stride in halves (80B: 16B aligned, bank-friendly)

__global__ __launch_bounds__(256)
void gemm_bf16f32(const float* __restrict__ A, const float* __restrict__ B,
                  float* __restrict__ C, int K, int N) {
  __shared__ bf16 As[128 * GSTR];
  __shared__ bf16 Bs[128 * GSTR];

  const int t = threadIdx.x;
  const int wave = t >> 5, lane = t & 31;
  const int wm = wave >> 1;       // 0..3 -> +32*wm rows
  const int wn = wave & 1;        // 0..1 -> +64*wn cols
  const int rowbase = blockIdx.y * 128;
  const int colbase = blockIdx.x * 128;
  const int khalf = (lane >> 4) * 8;
  const int ksel  = (lane >> 4) * 16;
  const int lm    = lane & 15;

  v8f acc[2][4];
#pragma unroll
  for (int i = 0; i < 2; ++i)
#pragma unroll
    for (int j = 0; j < 4; ++j) acc[i][j] = vzero();

  for (int k0 = 0; k0 < K; k0 += 32) {
    __syncthreads();
#pragma unroll
    for (int i = 0; i < 4; ++i) {
      int chunk = t + 256 * i;          // 1024 chunks of 4 floats
      int r  = chunk >> 3;              // 0..127
      int kk = (chunk & 7) * 4;         // 0..28
      const float* ap = A + (size_t)(rowbase + r) * K + k0 + kk;
      const float* bp = B + (size_t)(colbase + r) * K + k0 + kk;
      float4 av = *(const float4*)ap;
      float4 bv = *(const float4*)bp;
      if (k0 + 32 < K) {                // gfx1250 global_prefetch of next K tile
        __builtin_prefetch(ap + 32, 0, 3);
        __builtin_prefetch(bp + 32, 0, 3);
      }
      uint2 apk, bpk;
      apk.x = pk2(av.x, av.y); apk.y = pk2(av.z, av.w);
      bpk.x = pk2(bv.x, bv.y); bpk.y = pk2(bv.z, bv.w);
      *(uint2*)&As[r * GSTR + kk] = apk;
      *(uint2*)&Bs[r * GSTR + kk] = bpk;
    }
    __syncthreads();

    v16bf afr[2], bfr[4];
#pragma unroll
    for (int i = 0; i < 2; ++i)
      afr[i] = load_a_frag(&As[(wm * 32 + i * 16 + lm) * GSTR + ksel]);
#pragma unroll
    for (int j = 0; j < 4; ++j)
      bfr[j] = load_b_frag(&Bs[(wn * 64 + j * 16 + lm) * GSTR + ksel]);
#pragma unroll
    for (int i = 0; i < 2; ++i)
#pragma unroll
      for (int j = 0; j < 4; ++j)
        acc[i][j] = wmma_bf16(afr[i], bfr[j], acc[i][j]);
  }

  const int rowhi = (lane >> 4) * 8;
#pragma unroll
  for (int i = 0; i < 2; ++i)
#pragma unroll
    for (int j = 0; j < 4; ++j)
#pragma unroll
      for (int r = 0; r < 8; ++r) {
        int row = rowbase + wm * 32 + i * 16 + r + rowhi;
        int col = colbase + wn * 64 + j * 16 + lm;
        C[(size_t)row * N + col] = acc[i][j][r];
      }
}

// ---------------------------------------------------------------------------
// Kernel 2/4: per-(s, head-row) RMSNorm + RoPE, f32 -> bf16, head-major out.
// grid = (48, S), block = 128 (one thread per d).
// ---------------------------------------------------------------------------
__global__ __launch_bounds__(128)
void qkv_norm_rope(const float* __restrict__ qkv, const int* __restrict__ pos,
                   const float* __restrict__ gq, const float* __restrict__ gk,
                   bf16* __restrict__ Qb, bf16* __restrict__ Kb,
                   bf16* __restrict__ Vb) {
  const int idx = blockIdx.x;   // 0..47 head-row within qkv
  const int s   = blockIdx.y;
  const int d   = threadIdx.x;  // 0..127
  const float x = qkv[(size_t)s * QKV_N + idx * DHEAD + d];

  if (idx >= NH + NKV) {        // V rows: plain bf16 convert
    int kv = idx - (NH + NKV);
    Vb[((size_t)kv * S_LEN + s) * DHEAD + d] = (bf16)x;
    return;
  }

  __shared__ float xn[DHEAD];
  __shared__ float red[4];
  const int lane = threadIdx.x & 31, wv = threadIdx.x >> 5;

  float ss = x * x;
#pragma unroll
  for (int m = 1; m < 32; m <<= 1) ss += __shfl_xor(ss, m);
  if (lane == 0) red[wv] = ss;
  __syncthreads();
  float tot = red[0] + red[1] + red[2] + red[3];
  float inv = rsqrtf(tot * (1.0f / DHEAD) + EPSF);
  const float* g = (idx < NH) ? gq : gk;
  float y = x * inv * g[d];
  xn[d] = y;
  __syncthreads();

  int   i  = d & 63;
  float fr = (float)pos[s] * __expf(-(2.0f * i / DHEAD) * LN_THETA);
  float c  = __cosf(fr), sn = __sinf(fr);
  float other = xn[(d < 64) ? d + 64 : d - 64];
  float out   = (d < 64) ? (y * c - other * sn) : (y * c + other * sn);

  if (idx < NH) Qb[((size_t)idx * S_LEN + s) * DHEAD + d] = (bf16)out;
  else          Kb[((size_t)(idx - NH) * S_LEN + s) * DHEAD + d] = (bf16)out;
}

// ---------------------------------------------------------------------------
// Kernel 3/4: causal flash attention with GQA.
// grid = (S/64, H); block = 128 (4 waves). Wave w owns query rows 16w..16w+15.
// Q and K tiles are staged with async global->LDS (ASYNCcnt) copies.
// ---------------------------------------------------------------------------
#define QSTR 136   // 272B rows: 16B aligned
#define VSTR 72    // 144B rows
#define PSTR 72

__global__ __launch_bounds__(128)
void attn_kernel(const bf16* __restrict__ Qb, const bf16* __restrict__ Kb,
                 const bf16* __restrict__ Vb, float* __restrict__ ctx) {
  __shared__ bf16 Qs[64 * QSTR];
  __shared__ bf16 Ks[64 * QSTR];
  __shared__ bf16 Vt[DHEAD * VSTR];       // transposed: [d][key]
  __shared__ bf16 Ps[4 * 16 * PSTR];      // per-wave P tiles

  const int qt = blockIdx.x, h = blockIdx.y;
  const int kvh = h >> 2;                 // H/KV = 4
  const int t = threadIdx.x, wave = t >> 5, lane = t & 31;
  const int lm = lane & 15;
  const int khalf = (lane >> 4) * 8;
  const int ksel  = (lane >> 4) * 16;
  const int rowhi = (lane >> 4) * 8;
  const float scale = 0.08838834764831845f;   // 1/sqrt(128)

  // Load Q tile (64 x 128 bf16) via async global->LDS DMA
  const bf16* Qg = Qb + ((size_t)h * S_LEN + qt * 64) * DHEAD;
#pragma unroll
  for (int i = 0; i < 8; ++i) {
    int c = t + 128 * i;
    int r = c >> 4, off = (c & 15) * 8;
    async_g2l_b128(&Qs[r * QSTR + off], Qg + r * DHEAD + off);
  }

  v8f o[8];
  float mrow[8], lrow[8];
#pragma unroll
  for (int r = 0; r < 8; ++r) { mrow[r] = -3.0e38f; lrow[r] = 0.f; }
#pragma unroll
  for (int i = 0; i < 8; ++i) o[i] = vzero();

  for (int kt = 0; kt <= qt; ++kt) {
    __syncthreads();  // protect Ks/Vt against previous iteration readers
    const bf16* Kg = Kb + ((size_t)kvh * S_LEN + kt * 64) * DHEAD;
    const bf16* Vg = Vb + ((size_t)kvh * S_LEN + kt * 64) * DHEAD;
#pragma unroll
    for (int i = 0; i < 8; ++i) {
      int c = t + 128 * i;
      int r = c >> 4, off = (c & 15) * 8;
      // K tile: async global->LDS copy
      async_g2l_b128(&Ks[r * QSTR + off], Kg + r * DHEAD + off);
      // V: load 8 halves of a key row, scatter transposed into Vt[d][key]
      uint4 v = *(const uint4*)(Vg + r * DHEAD + off);
      unsigned int w0 = v.x, w1 = v.y, w2 = v.z, w3 = v.w;
      Vt[(off + 0) * VSTR + r] = us2bf((unsigned short)(w0 & 0xffff));
      Vt[(off + 1) * VSTR + r] = us2bf((unsigned short)(w0 >> 16));
      Vt[(off + 2) * VSTR + r] = us2bf((unsigned short)(w1 & 0xffff));
      Vt[(off + 3) * VSTR + r] = us2bf((unsigned short)(w1 >> 16));
      Vt[(off + 4) * VSTR + r] = us2bf((unsigned short)(w2 & 0xffff));
      Vt[(off + 5) * VSTR + r] = us2bf((unsigned short)(w2 >> 16));
      Vt[(off + 6) * VSTR + r] = us2bf((unsigned short)(w3 & 0xffff));
      Vt[(off + 7) * VSTR + r] = us2bf((unsigned short)(w3 >> 16));
    }
    wait_asynccnt0();   // Q (first iter) + K async copies landed in LDS
    __syncthreads();

    // S = Q * K^T : 16x64 per wave (4 accumulators), K-dim = 128 (4 steps)
    v8f sacc[4];
#pragma unroll
    for (int nt = 0; nt < 4; ++nt) sacc[nt] = vzero();
#pragma unroll
    for (int ks = 0; ks < 4; ++ks) {
      v16bf a = load_a_frag(&Qs[(wave * 16 + lm) * QSTR + ks * 32 + ksel]);
#pragma unroll
      for (int nt = 0; nt < 4; ++nt) {
        v16bf b = load_b_frag(&Ks[(nt * 16 + lm) * QSTR + ks * 32 + ksel]);
        sacc[nt] = wmma_bf16(a, b, sacc[nt]);
      }
    }

    // scale + causal mask (only the diagonal tile needs masking)
    const bool diag = (kt == qt);
#pragma unroll
    for (int nt = 0; nt < 4; ++nt)
#pragma unroll
      for (int r = 0; r < 8; ++r) {
        float sv = sacc[nt][r] * scale;
        if (diag) {
          int qrow = wave * 16 + r + rowhi;
          int kcol = nt * 16 + lm;
          if (kcol > qrow) sv = -3.0e38f;
        }
        sacc[nt][r] = sv;
      }

    // online softmax (rows live in (VGPR r, 16-lane half) of C layout)
#pragma unroll
    for (int r = 0; r < 8; ++r) {
      float mx = fmaxf(fmaxf(sacc[0][r], sacc[1][r]),
                       fmaxf(sacc[2][r], sacc[3][r]));
      mx = fmaxf(mx, __shfl_xor(mx, 1));
      mx = fmaxf(mx, __shfl_xor(mx, 2));
      mx = fmaxf(mx, __shfl_xor(mx, 4));
      mx = fmaxf(mx, __shfl_xor(mx, 8));
      float mnew  = fmaxf(mrow[r], mx);
      float alpha = __expf(mrow[r] - mnew);
      mrow[r] = mnew;
      float rs = 0.f;
#pragma unroll
      for (int nt = 0; nt < 4; ++nt) {
        float p = __expf(sacc[nt][r] - mnew);
        sacc[nt][r] = p;
        rs += p;
      }
      rs += __shfl_xor(rs, 1);
      rs += __shfl_xor(rs, 2);
      rs += __shfl_xor(rs, 4);
      rs += __shfl_xor(rs, 8);
      lrow[r] = lrow[r] * alpha + rs;
#pragma unroll
      for (int j = 0; j < 8; ++j) o[j][r] *= alpha;
    }

    // stage P (bf16) through per-wave LDS to get A-fragment layout
    bf16* Pw = &Ps[wave * 16 * PSTR];
#pragma unroll
    for (int nt = 0; nt < 4; ++nt)
#pragma unroll
      for (int r = 0; r < 8; ++r)
        Pw[(r + rowhi) * PSTR + nt * 16 + lm] = (bf16)sacc[nt][r];
    // same-wave LDS RAW; hardware DScnt wait inserted by compiler

    // O += P(16x64) * V(64x128)
#pragma unroll
    for (int ks = 0; ks < 2; ++ks) {
      v16bf a = load_a_frag(&Pw[lm * PSTR + ks * 32 + ksel]);
#pragma unroll
      for (int j = 0; j < 8; ++j) {
        v16bf b = load_b_frag(&Vt[(j * 16 + lm) * VSTR + ks * 32 + ksel]);
        o[j] = wmma_bf16(a, b, o[j]);
      }
    }
  }

  // normalize and write ctx[s, h*128 + d] (f32)
  float* Cg = ctx + (size_t)(qt * 64 + wave * 16) * (NH * DHEAD) + h * DHEAD;
#pragma unroll
  for (int r = 0; r < 8; ++r) {
    float inv = 1.0f / lrow[r];
    int row = r + rowhi;
#pragma unroll
    for (int j = 0; j < 8; ++j)
      Cg[(size_t)row * (NH * DHEAD) + j * 16 + lm] = o[j][r] * inv;
  }
}

// ---------------------------------------------------------------------------
// Launch
// ---------------------------------------------------------------------------
extern "C" void kernel_launch(void* const* d_in, const int* in_sizes, int n_in,
                              void* d_out, int out_size, void* d_ws,
                              size_t ws_size, hipStream_t stream) {
  const int*   positions = (const int*)d_in[0];
  const float* hs        = (const float*)d_in[1];
  const float* w_qkv     = (const float*)d_in[2];
  const float* w_o       = (const float*)d_in[3];
  const float* gq        = (const float*)d_in[4];
  const float* gk        = (const float*)d_in[5];
  float*       out       = (float*)d_out;

  char* ws = (char*)d_ws;
  float* qkv = (float*)ws;  ws += (size_t)S_LEN * QKV_N * sizeof(float);
  bf16*  Qb  = (bf16*)ws;   ws += (size_t)NH  * S_LEN * DHEAD * sizeof(bf16);
  bf16*  Kb  = (bf16*)ws;   ws += (size_t)NKV * S_LEN * DHEAD * sizeof(bf16);
  bf16*  Vb  = (bf16*)ws;   ws += (size_t)NKV * S_LEN * DHEAD * sizeof(bf16);
  float* ctx = (float*)ws;

  // 1) qkv = hs @ w_qkv^T        (2048 x 6144, K=4096)
  gemm_bf16f32<<<dim3(QKV_N / 128, S_LEN / 128), 256, 0, stream>>>(
      hs, w_qkv, qkv, HID, QKV_N);

  // 2) RMSNorm + RoPE, split into head-major bf16 Q/K/V
  qkv_norm_rope<<<dim3(NH + 2 * NKV, S_LEN), 128, 0, stream>>>(
      qkv, positions, gq, gk, Qb, Kb, Vb);

  // 3) causal flash attention -> ctx (2048 x 4096 f32)
  attn_kernel<<<dim3(S_LEN / 64, NH), 128, 0, stream>>>(Qb, Kb, Vb, ctx);

  // 4) out = ctx @ w_o^T         (2048 x 4096, K=4096)
  gemm_bf16f32<<<dim3(HID / 128, S_LEN / 128), 256, 0, stream>>>(
      ctx, w_o, out, HID, HID);
}
